// BilinearAttention_84756884619467
// MI455X (gfx1250) — matrix operand order repaired and checked by reading, not compiled, not run on hardware
//
#include <hip/hip_runtime.h>
#include <hip/hip_bf16.h>

// ---------------------------------------------------------------------------
// BilinearAttention on MI455X (gfx1250, wave32).
// Heavy math = two bf16 WMMA GEMMs (proj K=512, out K=128) + L2-resident edge
// gather. Tile staging into LDS uses the Tensor Data Mover (tensor_load_to_lds
// + s_wait_tensorcnt) where the toolchain exposes it; falls back to manual
// global->LDS staging otherwise.
// ---------------------------------------------------------------------------

#define NNODES 50000
#define DIN    512
#define DEG    32
#define NEDGE  (NNODES * DEG)

typedef __attribute__((ext_vector_type(16))) __bf16 v16bf;
typedef __attribute__((ext_vector_type(8)))  float  v8f;
typedef __attribute__((ext_vector_type(4)))  unsigned int v4u;
typedef __attribute__((ext_vector_type(8)))  int    v8i_;
typedef __attribute__((ext_vector_type(4)))  int    v4i_;

union BF16Frag {
    v16bf v;
    unsigned int u[8];
};

__device__ __forceinline__ unsigned int f2bf(float f) {
    unsigned int u = __float_as_uint(f);
    return ((u + 0x7FFFu + ((u >> 16) & 1u)) >> 16) & 0xFFFFu;  // RNE
}
__device__ __forceinline__ unsigned int pk2bf(float lo, float hi) {
    return f2bf(lo) | (f2bf(hi) << 16);
}
__device__ __forceinline__ float sigm(float w) { return 1.0f / (1.0f + __expf(-w)); }
__device__ __forceinline__ float nonneg(float w) { return w > 0.0f ? w + 1.0f : __expf(w); }  // elu(w)+1

// --------------------------------------------------------------- TDM load --
// 2D tile load Global->LDS via Tensor Data Mover (ISA ch.8 D# layout).
// data_size code 2 == 4-byte elements. Issued once per calling wave.
#define HAVE_TDM (__has_builtin(__builtin_amdgcn_tensor_load_to_lds) && \
                  __has_builtin(__builtin_amdgcn_s_wait_tensorcnt))

#if HAVE_TDM
__device__ __forceinline__ void tdm_load_2d_f32(unsigned int lds_off,
                                                const float* gaddr,
                                                unsigned int dim0,      // row len (elems)
                                                unsigned int dim1,      // total rows
                                                unsigned int tile0,     // tile row len
                                                unsigned int tile1,     // tile rows
                                                unsigned int stride0) { // row stride (elems)
    unsigned long long ga = (unsigned long long)(const void*)gaddr;
    v4u g0;
    g0[0] = 1u;                                           // count=1, user mode
    g0[1] = lds_off;                                      // lds_addr (bytes)
    g0[2] = (unsigned int)(ga & 0xFFFFFFFFu);             // global_addr[31:0]
    g0[3] = (unsigned int)((ga >> 32) & 0x01FFFFFFu)      // global_addr[56:32]
            | (2u << 30);                                 // type=2 ("image")
    v8i_ g1;
    g1[0] = (int)(2u << 16);                              // data_size=4B, mask=0
    g1[1] = (int)((dim0 & 0xFFFFu) << 16);                // tensor_dim0[15:0]
    g1[2] = (int)(((dim0 >> 16) & 0xFFFFu) |              // tensor_dim0[31:16]
                  ((dim1 & 0xFFFFu) << 16));              // tensor_dim1[15:0]
    g1[3] = (int)(((dim1 >> 16) & 0xFFFFu) |              // tensor_dim1[31:16]
                  ((tile0 & 0xFFFFu) << 16));             // tile_dim0
    g1[4] = (int)(tile1 & 0xFFFFu);                       // tile_dim1 (tile_dim2=0)
    g1[5] = (int)stride0;                                 // tensor_dim0_stride[31:0]
    g1[6] = 0;                                            // stride0[47:32] | stride1 lo
    g1[7] = 0;
    v4i_ z4 = {0, 0, 0, 0};
#if __clang_major__ >= 23
    v8i_ z8 = {0, 0, 0, 0, 0, 0, 0, 0};
    __builtin_amdgcn_tensor_load_to_lds(g0, g1, z4, z4, z8, 0);
#else
    __builtin_amdgcn_tensor_load_to_lds(g0, g1, z4, z4, 0);
#endif
}
#endif

__device__ __forceinline__ unsigned int lds_offset_of(const void* p) {
    // flat shared address = {SHARED_BASE, lds_offset}; low 32 bits is the offset
    return (unsigned int)(unsigned long long)p;
}

// -------------------------------------------------------------------- init --
__global__ void k_init(float* __restrict__ sumsc, float* __restrict__ xbar) {
    int i = blockIdx.x * 256 + threadIdx.x;
    if (i < DIN) xbar[i] = 0.0f;
    if (i < NNODES) sumsc[i] = 0.001f;
}

// ---------------------------------------------------- weight transforms ----
// W_all [192][512] bf16: rows 0-31 w_ego (plain), 32-95 normed(q_local),
//                        96-159 nonneg(k_local), 160-191 normed(q_global)
// V_all [512][128] bf16: cols 0-31 nonneg(v_ego), 32-95 nonneg(v_local),
//                        96-127 nonneg(v_global)
__global__ __launch_bounds__(1024) void k_wprep(
        const float* __restrict__ wego, const float* __restrict__ vego,
        const float* __restrict__ qlw,  const float* __restrict__ klw,
        const float* __restrict__ vlw,  const float* __restrict__ qgw,
        const float* __restrict__ vgw,  const float* __restrict__ bias,
        unsigned short* __restrict__ Wall, unsigned short* __restrict__ Vall,
        float* __restrict__ biasf) {
    __shared__ float red[1024];
    const int tid = threadIdx.x;

    float s0 = 0.0f, s1 = 0.0f;
    for (int i = tid; i < 64 * DIN; i += 1024) s0 += sigm(qlw[i]);
    for (int i = tid; i < 32 * DIN; i += 1024) s1 += sigm(qgw[i]);

    red[tid] = s0; __syncthreads();
    for (int off = 512; off > 0; off >>= 1) {
        if (tid < off) red[tid] += red[tid + off];
        __syncthreads();
    }
    const float rSql = 1.0f / red[0];
    __syncthreads();

    red[tid] = s1; __syncthreads();
    for (int off = 512; off > 0; off >>= 1) {
        if (tid < off) red[tid] += red[tid + off];
        __syncthreads();
    }
    const float rSqg = 1.0f / red[0];
    __syncthreads();

    for (int i = tid; i < 32 * DIN; i += 1024) Wall[i]             = (unsigned short)f2bf(wego[i]);
    for (int i = tid; i < 64 * DIN; i += 1024) Wall[32 * DIN + i]  = (unsigned short)f2bf(sigm(qlw[i]) * rSql);
    for (int i = tid; i < 64 * DIN; i += 1024) Wall[96 * DIN + i]  = (unsigned short)f2bf(nonneg(klw[i]));
    for (int i = tid; i < 32 * DIN; i += 1024) Wall[160 * DIN + i] = (unsigned short)f2bf(sigm(qgw[i]) * rSqg);

    for (int i = tid; i < 512 * 128; i += 1024) {
        int o = i >> 7, j = i & 127;
        float w;
        if (j < 32)      w = vego[o * 32 + j];
        else if (j < 96) w = vlw[o * 64 + (j - 32)];
        else             w = vgw[o * 32 + (j - 96)];
        Vall[(size_t)o * 128 + j] = (unsigned short)f2bf(nonneg(w));
    }
    for (int i = tid; i < DIN; i += 1024) biasf[i] = nonneg(bias[i]);
}

// -------------------------------------------------------- x_bar reduction --
__global__ __launch_bounds__(256) void k_xbar(const float* __restrict__ x,
                                              float* __restrict__ xbar) {
    const int b = blockIdx.x, tid = threadIdx.x;
    const int c0 = tid * 2;
    float a0 = 0.0f, a1 = 0.0f;
    const int rbeg = b * 250, rend = rbeg + 250;   // 200 blocks * 250 rows = 50000
    for (int r = rbeg; r < rend; ++r) {
        const float* xr = x + (size_t)r * DIN + c0;
        a0 += xr[0]; a1 += xr[1];
    }
    atomicAdd(&xbar[c0], a0);
    atomicAdd(&xbar[c0 + 1], a1);
}

// --------------------------------------------------- k_g = xbar @ nonneg.T --
__global__ __launch_bounds__(256) void k_kg(const float* __restrict__ xbarsum,
                                            const float* __restrict__ kgw,
                                            float* __restrict__ kg) {
    const int wave = threadIdx.x >> 5, lane = threadIdx.x & 31;
    for (int j = wave; j < 32; j += 8) {
        float s = 0.0f;
        for (int i = lane; i < DIN; i += 32)
            s += (xbarsum[i] * (1.0f / (float)NNODES)) * nonneg(kgw[j * DIN + i]);
        for (int off = 16; off; off >>= 1) s += __shfl_xor(s, off, 32);
        if (lane == 0) kg[j] = s;
    }
}

__device__ __forceinline__ void rowsum_atomic(float val, int row, int m,
                                              float* __restrict__ sumsc) {
    for (int off = 8; off; off >>= 1) val += __shfl_xor(val, off, 16);
    if (m == 0) atomicAdd(&sumsc[row], val);
}

// ------------------------------------------- projection GEMM (K=512, bf16) --
// 16-row tile per block; 12 waves = 12 output col-tiles (192 cols).
// x tile staged via TDM (fp32, 32KB), packed to bf16 (16KB) once, then the
// WMMA K-loop reads A from LDS and streams B from global (L2-hot weights).
__global__ __launch_bounds__(384) void k_proj(
        const float* __restrict__ x, const unsigned short* __restrict__ Wall,
        const float* __restrict__ kg, float* __restrict__ projqk,
        float* __restrict__ proj2, float* __restrict__ sumsc) {
    __shared__ float        ldsXf[16 * DIN];   // staged fp32 tile (32KB)
    __shared__ unsigned int ldsA[16 * 256];    // packed bf16 tile (16KB)
    const int row0 = blockIdx.x * 16;
    const int tid = threadIdx.x;

#if HAVE_TDM
    if (tid < 32) {   // wave 0 issues one DMA; other waves branch around it
        tdm_load_2d_f32(lds_offset_of(ldsXf), x + (size_t)row0 * DIN,
                        DIN, NNODES, DIN, 16, DIN);
        __builtin_amdgcn_s_wait_tensorcnt(0);
    }
    __syncthreads();
#else
    for (int p = tid; p < 16 * DIN; p += 384)
        ldsXf[p] = x[(size_t)row0 * DIN + p];
    __syncthreads();
#endif
    for (int p = tid; p < 16 * 256; p += 384)
        ldsA[p] = pk2bf(ldsXf[p * 2], ldsXf[p * 2 + 1]);
    __syncthreads();

    const int wave = tid >> 5, lane = tid & 31;
    const int h = lane >> 4, m = lane & 15;
    const int n0 = wave * 16, nl = n0 + m;

    v8f c = {0, 0, 0, 0, 0, 0, 0, 0};
    for (int kk = 0; kk < 16; ++kk) {
        const int k0 = kk * 32;
        BF16Frag a, b;
        // A (16x32 bf16): half-wave splits K 0-15 / 16-31 (contiguous, matches B)
        #pragma unroll
        for (int i = 0; i < 8; ++i) {
            const int grp = i >> 2, j = i & 3;
            const int kb = k0 + h * 16 + (grp * 4 + j) * 2;
            a.u[i] = ldsA[(m << 8) + (kb >> 1)];
        }
        // B (32x16 bf16): lane -> column nl, half-wave splits K 0-15 / 16-31
        const unsigned int* wrow =
            (const unsigned int*)(Wall + (size_t)nl * DIN + k0 + h * 16);
        #pragma unroll
        for (int i = 0; i < 8; ++i) b.u[i] = wrow[i];

        c = __builtin_amdgcn_wmma_f32_16x16x32_bf16(false, a.v, false, b.v,
                                                    (short)0, c, false, false);
    }

    const float inv512 = 1.0f / 512.0f;
    const float inv = 1.0f / (512.0f * 512.0f);

    if (wave < 2) {                       // ego: square, store, row-sum
        #pragma unroll
        for (int r = 0; r < 8; ++r) {
            const int row = row0 + r + 8 * h;
            const float e = c[r] * inv512;
            const float s = e * e;
            proj2[(size_t)row * 128 + nl] = s;
            rowsum_atomic(s, row, m, sumsc);
        }
    } else if (wave < 10) {               // q_l / k_l raw for edge gather
        const int col = nl - 32;          // 0-63 q, 64-127 k
        #pragma unroll
        for (int r = 0; r < 8; ++r) {
            const int row = row0 + r + 8 * h;
            projqk[(size_t)row * 128 + col] = c[r];
        }
    } else {                              // global: * k_g * inv, store, row-sum
        const int j = nl - 160;
        const float kgj = kg[j] * inv;
        #pragma unroll
        for (int r = 0; r < 8; ++r) {
            const int row = row0 + r + 8 * h;
            const float g = c[r] * kgj;
            proj2[(size_t)row * 128 + 96 + j] = g;
            rowsum_atomic(g, row, m, sumsc);
        }
    }
}

// ------------------------------------------------------------ edge gather --
// one wave per node: 32 edges x 64 dims (float2 per lane); q_l/k_l are
// 12.8 MB each -> gathers hit L2 (192 MB).
__global__ __launch_bounds__(256) void k_edge(const int* __restrict__ adj,
                                              const float* __restrict__ projqk,
                                              float* __restrict__ proj2,
                                              float* __restrict__ sumsc) {
    const int wave = threadIdx.x >> 5, lane = threadIdx.x & 31;
    const int node = blockIdx.x * 8 + wave;
    const float inv = 1.0f / (512.0f * 512.0f);
    float ax = 0.0f, ay = 0.0f;
    const int ebase = node * DEG;
    for (int i = 0; i < DEG; ++i) {
        const int a0 = adj[ebase + i];          // src
        const int a1 = adj[NEDGE + ebase + i];  // dst
        const float* q = projqk + (size_t)a1 * 128 + lane * 2;
        const float* k = projqk + (size_t)a0 * 128 + 64 + lane * 2;
        ax += q[0] * k[0];
        ay += q[1] * k[1];
    }
    ax *= inv; ay *= inv;
    float2 st; st.x = ax; st.y = ay;
    *(float2*)(proj2 + (size_t)node * 128 + 32 + lane * 2) = st;
    float s = ax + ay;
    for (int off = 16; off; off >>= 1) s += __shfl_xor(s, off, 32);
    if (lane == 0) sumsc[node] += s;   // exclusive writer; kernels are ordered
}

// ----------------------------------------------- output GEMM (K=128, bf16) --
// 16-row tile; 8 waves x 4 col-tiles = 512 outputs. Scores tile staged via
// TDM into LDS; A fragments (scores/sum, bf16) built once per wave and
// reused across the 4 col-tiles.
__global__ __launch_bounds__(256) void k_out(const float* __restrict__ proj2,
                                             const float* __restrict__ sumsc,
                                             const unsigned short* __restrict__ Vall,
                                             const float* __restrict__ biasf,
                                             float* __restrict__ out) {
    __shared__ float ldsP[16 * 128];   // 8KB score tile
    const int row0 = blockIdx.x * 16;
    const int tid = threadIdx.x;
    const int wave = tid >> 5, lane = tid & 31;
    const int h = lane >> 4, m = lane & 15;

#if HAVE_TDM
    if (tid < 32) {
        tdm_load_2d_f32(lds_offset_of(ldsP), proj2 + (size_t)row0 * 128,
                        128, NNODES, 128, 16, 128);
        __builtin_amdgcn_s_wait_tensorcnt(0);
    }
    __syncthreads();
#else
    for (int p = tid; p < 16 * 128; p += 256)
        ldsP[p] = proj2[(size_t)row0 * 128 + p];
    __syncthreads();
#endif

    const float scale = 1.0f / sumsc[row0 + m];
    const float* prow = ldsP + m * 128;

    BF16Frag a[4];
    #pragma unroll
    for (int kk = 0; kk < 4; ++kk) {
        #pragma unroll
        for (int i = 0; i < 8; ++i) {
            const int grp = i >> 2, j = i & 3;
            const int kb = kk * 32 + h * 16 + (grp * 4 + j) * 2;
            a[kk].u[i] = pk2bf(prow[kb] * scale, prow[kb + 1] * scale);
        }
    }

    for (int t = 0; t < 4; ++t) {
        const int n0 = (wave + t * 8) * 16;
        const int nl = n0 + m;
        v8f c = {0, 0, 0, 0, 0, 0, 0, 0};
        #pragma unroll
        for (int kk = 0; kk < 4; ++kk) {
            BF16Frag b;
            const unsigned int* vr =
                (const unsigned int*)(Vall + (size_t)nl * 128 + kk * 32 + h * 16);
            #pragma unroll
            for (int i = 0; i < 8; ++i) b.u[i] = vr[i];
            c = __builtin_amdgcn_wmma_f32_16x16x32_bf16(false, a[kk].v, false, b.v,
                                                        (short)0, c, false, false);
        }
        const float bn = biasf[nl];
        #pragma unroll
        for (int r = 0; r < 8; ++r)
            out[(size_t)(row0 + r + 8 * h) * DIN + nl] = c[r] + bn;
    }
}

// ---------------------------------------------------------------------------
extern "C" void kernel_launch(void* const* d_in, const int* in_sizes, int n_in,
                              void* d_out, int out_size, void* d_ws, size_t ws_size,
                              hipStream_t stream) {
    const int*   adj  = (const int*)d_in[0];
    const float* x    = (const float*)d_in[1];
    const float* wego = (const float*)d_in[2];
    const float* vego = (const float*)d_in[3];
    const float* qlw  = (const float*)d_in[4];
    const float* klw  = (const float*)d_in[5];
    const float* vlw  = (const float*)d_in[6];
    const float* qgw  = (const float*)d_in[7];
    const float* kgw  = (const float*)d_in[8];
    const float* vgw  = (const float*)d_in[9];
    const float* bias = (const float*)d_in[10];
    float* out = (float*)d_out;

    char* ws = (char*)d_ws;
    size_t o = 0;
    auto take = [&](size_t bytes) { size_t r = o; o += (bytes + 255) & ~(size_t)255; return r; };
    unsigned short* Wall  = (unsigned short*)(ws + take(192 * DIN * 2));
    unsigned short* Vall  = (unsigned short*)(ws + take(512 * 128 * 2));
    float*          biasf = (float*)(ws + take(DIN * 4));
    float*          kg    = (float*)(ws + take(32 * 4));
    float*          xbar  = (float*)(ws + take(DIN * 4));
    float*          sumsc = (float*)(ws + take((size_t)NNODES * 4));
    float*          pqk   = (float*)(ws + take((size_t)NNODES * 128 * 4));
    float*          p2    = (float*)(ws + take((size_t)NNODES * 128 * 4));
    (void)ws_size; (void)in_sizes; (void)n_in; (void)out_size;

    k_init <<<(NNODES + 255) / 256, 256, 0, stream>>>(sumsc, xbar);
    k_wprep<<<1, 1024, 0, stream>>>(wego, vego, qlw, klw, vlw, qgw, vgw, bias,
                                    Wall, Vall, biasf);
    k_xbar <<<200, 256, 0, stream>>>(x, xbar);
    k_kg   <<<1, 256, 0, stream>>>(xbar, kgw, kg);
    k_proj <<<NNODES / 16, 384, 0, stream>>>(x, Wall, kg, pqk, p2, sumsc);
    k_edge <<<NNODES / 8, 256, 0, stream>>>(adj, pqk, p2, sumsc);
    k_out  <<<NNODES / 16, 256, 0, stream>>>(p2, sumsc, Vall, biasf, out);
}
